// TripleSR_20366734917802
// MI455X (gfx1250) — compile-verified, matrix operand order checked by reading, not run on hardware
//
#include <hip/hip_runtime.h>
#include <hip/hip_bf16.h>

// ---------------------------------------------------------------------------
// TripleSR fused pipeline for gfx1250 (MI455X)
//   prep       : repack MLP + conv weights into WMMA B-fragment order (f16)
//   conv1      : 1->64ch 3x3x3 conv + relu, f16 channel-last out (tiny)
//   conv_wmma  : 64->64ch 3x3x3 conv as implicit GEMM via v_wmma (f32 acc)
//   fused      : trilinear gather -> LDS, then 6-layer MLP in LDS;
//                A fragments register-cached; B weights double-buffered in
//                LDS via GLOBAL_LOAD_ASYNC_TO_LDS_B128 (ASYNCcnt pipeline)
// ---------------------------------------------------------------------------

typedef __attribute__((ext_vector_type(16))) _Float16 v16h;
typedef __attribute__((ext_vector_type(8)))  _Float16 v8h;
typedef __attribute__((ext_vector_type(8)))  float    v8f;

#define F_DIM 64
#define GRID 32
#define NVOX (GRID*GRID*GRID)
#define P_TOTAL 262144
#define BM 64

struct PrepParams {
    const float* w1; const float* w2; const float* w3;
    const float* mw[6];
    const float* mb[6];
    _Float16* w1T;
    _Float16* pc2; _Float16* pc3;
    _Float16* wp;
    float* biasB;
};

// ---------------------------------------------------------------------------
// prep: grid-stride over all repack jobs
// ---------------------------------------------------------------------------
__global__ void prep_kernel(PrepParams P) {
    const int lsz[6]  = {32768, 65536, 65536, 65536, 65536, 4096};
    const int ndim[6] = {256, 256, 256, 256, 256, 1};
    const int ks[6]   = {4, 8, 8, 8, 8, 8};
    const long MLP_TOTAL = 299008;
    const long BIAS_END  = MLP_TOTAL + 1536;
    const long W1_END    = BIAS_END + 1728;
    const long PC2_END   = W1_END + 110592;
    const long PC3_END   = PC2_END + 110592;

    const long stride = (long)gridDim.x * blockDim.x;
    for (long id = (long)blockIdx.x * blockDim.x + threadIdx.x; id < PC3_END; id += stride) {
        if (id < MLP_TOTAL) {
            long r = id; int L = 0;
            while (L < 6 && r >= lsz[L]) { r -= lsz[L]; ++L; }
            const int per_nt = ks[L] * 512;
            const int nt  = (int)(r / per_nt);
            int rem = (int)(r % per_nt);
            const int kt  = rem / 512; rem &= 511;
            const int lane = rem >> 4;
            const int j    = rem & 15;
            const int n   = nt * 16 + (lane & 15);
            const int jlo = (j < 8) ? j : (j + 8);
            const int k   = kt * 32 + ((lane >= 16) ? 8 : 0) + jlo;
            float v = 0.f;
            if (n < ndim[L]) v = P.mw[L][(long)k * ndim[L] + n];
            P.wp[id] = (_Float16)v;
        } else if (id < BIAS_END) {
            const int r = (int)(id - MLP_TOTAL);
            const int L = r >> 8, n = r & 255;
            P.biasB[r] = (n < ndim[L]) ? P.mb[L][n] : 0.f;
        } else if (id < W1_END) {
            const int r = (int)(id - BIAS_END);
            const int t = r >> 6, co = r & 63;
            P.w1T[r] = (_Float16)P.w1[co * 27 + t];
        } else {
            const int which = (id >= PC2_END) ? 1 : 0;         // 0=pc2, 1=pc3
            const int r = (int)(id - (which ? PC2_END : W1_END));
            const int per_nt = 54 * 512;
            const int nt  = r / per_nt;
            int rem = r % per_nt;
            const int kt  = rem / 512; rem &= 511;
            const int lane = rem >> 4;
            const int j    = rem & 15;
            const int co  = nt * 16 + (lane & 15);
            const int jlo = (j < 8) ? j : (j + 8);
            const int k   = kt * 32 + ((lane >= 16) ? 8 : 0) + jlo;
            const int tap = k >> 6, ci = k & 63;
            const float* w = which ? P.w3 : P.w2;
            const float v = w[(long)co * 1728 + ci * 27 + tap];
            (which ? P.pc3 : P.pc2)[r] = (_Float16)v;
        }
    }
}

// ---------------------------------------------------------------------------
// conv1: (1ch -> 64ch) + relu, f16 channel-last out
// ---------------------------------------------------------------------------
__global__ __launch_bounds__(64) void conv1_relu(const float* __restrict__ x,
                                                 const _Float16* __restrict__ w1T,
                                                 const float* __restrict__ b1,
                                                 _Float16* __restrict__ out) {
    const int v  = blockIdx.x;
    const int co = threadIdx.x;
    const int d = v >> 10, h = (v >> 5) & 31, w = v & 31;
    float acc = b1[co];
    #pragma unroll
    for (int t = 0; t < 27; ++t) {
        const int dz = t / 9 - 1, dy = (t / 3) % 3 - 1, dx = t % 3 - 1;
        const int zz = d + dz, yy = h + dy, xx = w + dx;
        if ((unsigned)zz < 32u && (unsigned)yy < 32u && (unsigned)xx < 32u)
            acc = fmaf(x[(zz * 32 + yy) * 32 + xx], (float)w1T[t * 64 + co], acc);
    }
    out[(long)v * 64 + co] = (_Float16)(acc > 0.f ? acc : 0.f);
}

// ---------------------------------------------------------------------------
// conv_wmma: 64->64ch 3x3x3 conv, implicit GEMM (see round-2 notes)
// ---------------------------------------------------------------------------
__global__ __launch_bounds__(128) void conv_wmma(
        const _Float16* __restrict__ in,
        const _Float16* __restrict__ wpk,
        const float* __restrict__ bias,
        const _Float16* __restrict__ resid,
        _Float16* __restrict__ outH,
        float* __restrict__ outF,
        int mode) {
    const int nt   = threadIdx.x >> 5;
    const int lane = threadIdx.x & 31;
    const int vbase = blockIdx.x * 16;
    const int m = lane & 15;
    const int vox = vbase + m;
    const int x = vox & 31, y = (vox >> 5) & 31, z = vox >> 10;
    const int col  = nt * 16 + (lane & 15);
    const int hsel = (lane >> 4) << 3;

    v8f acc;
    const float bv = bias[col];
    #pragma unroll
    for (int i = 0; i < 8; ++i) acc[i] = bv;

    #pragma unroll 2
    for (int kt = 0; kt < 54; ++kt) {
        const int tap = kt >> 1;
        const int dz = tap / 9 - 1, dy = (tap / 3) % 3 - 1, dx = tap % 3 - 1;
        const int zz = z + dz, yy = y + dy, xx = x + dx;
        v16h a;
        if ((unsigned)zz < 32u && (unsigned)yy < 32u && (unsigned)xx < 32u) {
            const int ci0 = ((kt & 1) << 5) + hsel;
            const _Float16* ap = in + (long)((zz * 32 + yy) * 32 + xx) * 64 + ci0;
            const v8h a0 = *(const v8h*)ap;
            const v8h a1 = *(const v8h*)(ap + 16);
            #pragma unroll
            for (int i = 0; i < 8; ++i) { a[i] = a0[i]; a[i + 8] = a1[i]; }
        } else {
            #pragma unroll
            for (int i = 0; i < 16; ++i) a[i] = (_Float16)0.f;
        }
        const _Float16* bp = wpk + ((long)(nt * 54 + kt) * 32 + lane) * 16;
        const v8h b0 = *(const v8h*)bp;
        const v8h b1 = *(const v8h*)(bp + 8);
        v16h b;
        #pragma unroll
        for (int i = 0; i < 8; ++i) { b[i] = b0[i]; b[i + 8] = b1[i]; }
        acc = __builtin_amdgcn_wmma_f32_16x16x32_f16(false, a, false, b,
                                                     (short)0, acc, false, false);
    }
    #pragma unroll
    for (int i = 0; i < 8; ++i) {
        const int row  = i + ((lane >> 4) << 3);
        const long voxo = (long)(vbase + row);
        float v = acc[i];
        if (mode == 0) {
            v = v > 0.f ? v : 0.f;
            outH[voxo * 64 + col] = (_Float16)v;
        } else {
            v += (float)resid[voxo * 64 + col];
            outF[voxo * 64 + col] = v;
        }
    }
}

// ---------------------------------------------------------------------------
// async global -> LDS staging (ASYNCcnt path), 16B per lane per issue
// ---------------------------------------------------------------------------
__device__ __forceinline__ void stage_async(const _Float16* gsrc, _Float16* lds,
                                            int bytes, int tid) {
    const unsigned lbase = (unsigned)(unsigned long long)lds;
    const char* g = (const char*)gsrc;
    for (int off = tid * 16; off < bytes; off += 256 * 16) {
        unsigned laddr = lbase + (unsigned)off;
        unsigned long long gaddr = (unsigned long long)(g + off);
        asm volatile("global_load_async_to_lds_b128 %0, %1, off"
                     :: "v"(laddr), "v"(gaddr) : "memory");
    }
}

__device__ __forceinline__ void wait_async0() {
    asm volatile("s_wait_asynccnt 0x0" ::: "memory");
}

// ---------------------------------------------------------------------------
// fused gather + 6-layer MLP
// ---------------------------------------------------------------------------
__device__ __forceinline__ void axis_lerp(int X, int& i0, int& i1, float& f) {
    const float s  = 0.5f * (float)X - 0.25f;   // align_corners=False, 2x
    const float fl = floorf(s);
    f = s - fl;
    int i = (int)fl;
    i0 = i < 0 ? 0 : (i > 31 ? 31 : i);
    int ip = i + 1;
    i1 = ip < 0 ? 0 : (ip > 31 ? 31 : ip);
}

// One MLP layer. wave -> row-tile rt = wave/2, ntl = wave&1 picks one of the
// 2 N-tiles of the current staged chunk. A cached in registers for the layer;
// B double-buffered in LDS via async copies (issue next / compute current).
template <int KS, int NTC>
__device__ __forceinline__ void mlp_layer(const _Float16* inb, _Float16* outb,
                                          float* out, const _Float16* wpL,
                                          const float* biasL,
                                          _Float16* stg0, _Float16* stg1,
                                          int wave, int lane, int tid,
                                          long rowBase) {
    const int rt  = wave >> 1;
    const int ntl = wave & 1;
    const int r   = rt * 16 + (lane & 15);
    const int khb = (lane >> 4) << 3;

    v16h af[KS];
    #pragma unroll
    for (int kt = 0; kt < KS; ++kt) {
        const v8h a0 = *(const v8h*)&inb[r * 256 + kt * 32 + khb];
        const v8h a1 = *(const v8h*)&inb[r * 256 + kt * 32 + khb + 16];
        #pragma unroll
        for (int i = 0; i < 8; ++i) { af[kt][i] = a0[i]; af[kt][i + 8] = a1[i]; }
    }

    const int nchunks     = (NTC == 16) ? 8 : 1;
    const int chunkHalves = (NTC == 16 ? 2 : 1) * KS * 512;
    const int chunkBytes  = chunkHalves * 2;

    stage_async(wpL, stg0, chunkBytes, tid);     // prologue: chunk 0
    wait_async0();
    __syncthreads();

    for (int c = 0; c < nchunks; ++c) {
        const _Float16* cur = (c & 1) ? stg1 : stg0;
        if (c + 1 < nchunks)                      // overlap: stage next chunk
            stage_async(wpL + (long)(c + 1) * chunkHalves,
                        (c & 1) ? stg0 : stg1, chunkBytes, tid);

        const bool active = (NTC == 16) || (ntl == 0);
        if (active) {
            const int nt  = (NTC == 16) ? (c * 2 + ntl) : 0;
            const int col = nt * 16 + (lane & 15);
            const float bias = biasL[col];
            v8f acc;
            #pragma unroll
            for (int i = 0; i < 8; ++i) acc[i] = bias;
            const _Float16* bbase = cur + ((long)ntl * KS * 32 + lane) * 16;
            #pragma unroll
            for (int kt = 0; kt < KS; ++kt) {
                const v8h b0 = *(const v8h*)(bbase + (long)kt * 512);
                const v8h b1 = *(const v8h*)(bbase + (long)kt * 512 + 8);
                v16h b;
                #pragma unroll
                for (int i = 0; i < 8; ++i) { b[i] = b0[i]; b[i + 8] = b1[i]; }
                acc = __builtin_amdgcn_wmma_f32_16x16x32_f16(
                        false, af[kt], false, b, (short)0, acc, false, false);
            }
            if (NTC == 16) {
                #pragma unroll
                for (int i = 0; i < 8; ++i) {
                    float x = acc[i];
                    x = x > 0.f ? x : 0.f;
                    const int row = rt * 16 + i + ((lane >> 4) << 3);
                    outb[row * 256 + col] = (_Float16)x;
                }
            } else if ((lane & 15) == 0) {
                #pragma unroll
                for (int i = 0; i < 8; ++i) {
                    const int row = rt * 16 + i + ((lane >> 4) << 3);
                    out[rowBase + row] = acc[i];
                }
            }
        }
        wait_async0();                            // next chunk landed
        __syncthreads();
    }
}

__global__ __launch_bounds__(256) void fused_gather_mlp(
        const float* __restrict__ fa, const float* __restrict__ fc,
        const int* __restrict__ pix, const _Float16* __restrict__ wp,
        const float* __restrict__ biasB, float* __restrict__ out) {
    __shared__ _Float16 actA[BM * 256];
    __shared__ _Float16 actB[BM * 256];
    __shared__ __align__(16) _Float16 stg0[8192];   // 16 KB B-chunk buffers
    __shared__ __align__(16) _Float16 stg1[8192];

    const int tid  = threadIdx.x;
    const int wave = tid >> 5;
    const int lane = tid & 31;
    const long rowBase = (long)blockIdx.x * BM;

    // ---- on-the-fly trilinear gather: 64 points x 128 feats -> actA -------
    for (int it = 0; it < 32; ++it) {
        const int idx = it * 256 + tid;
        const int pl  = idx >> 7;
        const int c2  = idx & 127;
        const long p  = rowBase + pl;
        const int X = pix[p * 3 + 0], Y = pix[p * 3 + 1], Z = pix[p * 3 + 2];
        int iz0, iz1, iy0, iy1, ix0, ix1; float fz, fy, fx;
        axis_lerp(X, iz0, iz1, fz);
        axis_lerp(Y, iy0, iy1, fy);
        axis_lerp(Z, ix0, ix1, fx);
        const float* src = (c2 < 64) ? fa : fc;
        const int c = c2 & 63;
        const float a000 = src[((iz0 * 32 + iy0) * 32 + ix0) * 64 + c];
        const float a001 = src[((iz0 * 32 + iy0) * 32 + ix1) * 64 + c];
        const float a010 = src[((iz0 * 32 + iy1) * 32 + ix0) * 64 + c];
        const float a011 = src[((iz0 * 32 + iy1) * 32 + ix1) * 64 + c];
        const float a100 = src[((iz1 * 32 + iy0) * 32 + ix0) * 64 + c];
        const float a101 = src[((iz1 * 32 + iy0) * 32 + ix1) * 64 + c];
        const float a110 = src[((iz1 * 32 + iy1) * 32 + ix0) * 64 + c];
        const float a111 = src[((iz1 * 32 + iy1) * 32 + ix1) * 64 + c];
        const float v0 = (1.f - fy) * ((1.f - fx) * a000 + fx * a001)
                       +        fy  * ((1.f - fx) * a010 + fx * a011);
        const float v1 = (1.f - fy) * ((1.f - fx) * a100 + fx * a101)
                       +        fy  * ((1.f - fx) * a110 + fx * a111);
        actA[pl * 256 + c2] = (_Float16)((1.f - fz) * v0 + fz * v1);
    }
    __syncthreads();

    // ---- 6-layer MLP, ping-pong in LDS ------------------------------------
    mlp_layer<4, 16>(actA, actB, out, wp,          biasB,        stg0, stg1, wave, lane, tid, rowBase);
    __syncthreads();
    mlp_layer<8, 16>(actB, actA, out, wp + 32768,  biasB + 256,  stg0, stg1, wave, lane, tid, rowBase);
    __syncthreads();
    mlp_layer<8, 16>(actA, actB, out, wp + 98304,  biasB + 512,  stg0, stg1, wave, lane, tid, rowBase);
    __syncthreads();
    mlp_layer<8, 16>(actB, actA, out, wp + 163840, biasB + 768,  stg0, stg1, wave, lane, tid, rowBase);
    __syncthreads();
    mlp_layer<8, 16>(actA, actB, out, wp + 229376, biasB + 1024, stg0, stg1, wave, lane, tid, rowBase);
    __syncthreads();
    mlp_layer<8, 1>(actB, actA, out, wp + 294912,  biasB + 1280, stg0, stg1, wave, lane, tid, rowBase);
}

// ---------------------------------------------------------------------------
// host launch
// ---------------------------------------------------------------------------
extern "C" void kernel_launch(void* const* d_in, const int* in_sizes, int n_in,
                              void* d_out, int out_size, void* d_ws, size_t ws_size,
                              hipStream_t stream) {
    (void)in_sizes; (void)n_in; (void)out_size; (void)ws_size;
    const float* imgA = (const float*)d_in[0];
    const float* imgC = (const float*)d_in[1];
    const int*   pix  = (const int*)d_in[3];
    const float* ew1  = (const float*)d_in[5];
    const float* eb1  = (const float*)d_in[6];
    const float* ew2  = (const float*)d_in[7];
    const float* eb2  = (const float*)d_in[8];
    const float* ew3  = (const float*)d_in[9];
    const float* eb3  = (const float*)d_in[10];

    char* base = (char*)d_ws;
    size_t off = 0;
    auto carve = [&](size_t bytes) -> void* {
        void* p = base + off;
        off = (off + bytes + 255) & ~(size_t)255;
        return p;
    };
    float*    fa    = (float*)   carve((size_t)NVOX * 64 * 4);
    float*    fcv   = (float*)   carve((size_t)NVOX * 64 * 4);
    _Float16* tmpH  = (_Float16*)carve((size_t)NVOX * 64 * 2);
    _Float16* tmpF  = (_Float16*)carve((size_t)NVOX * 64 * 2);
    _Float16* w1T   = (_Float16*)carve(1728 * 2);
    _Float16* pc2   = (_Float16*)carve(110592 * 2);
    _Float16* pc3   = (_Float16*)carve(110592 * 2);
    float*    biasB = (float*)   carve(1536 * 4);
    _Float16* wp    = (_Float16*)carve(299008 * 2);

    PrepParams P;
    P.w1 = ew1; P.w2 = ew2; P.w3 = ew3;
    for (int i = 0; i < 6; ++i) {
        P.mw[i] = (const float*)d_in[11 + i];
        P.mb[i] = (const float*)d_in[17 + i];
    }
    P.w1T = w1T; P.pc2 = pc2; P.pc3 = pc3; P.wp = wp; P.biasB = biasB;

    prep_kernel<<<512, 256, 0, stream>>>(P);

    // encoder A: h = relu(conv1); f = relu(conv2(h)); fa = conv3(f) + h
    conv1_relu<<<NVOX, 64, 0, stream>>>(imgA, w1T, eb1, tmpH);
    conv_wmma<<<NVOX / 16, 128, 0, stream>>>(tmpH, pc2, eb2, nullptr, tmpF, nullptr, 0);
    conv_wmma<<<NVOX / 16, 128, 0, stream>>>(tmpF, pc3, eb3, tmpH, nullptr, fa, 1);
    // encoder C
    conv1_relu<<<NVOX, 64, 0, stream>>>(imgC, w1T, eb1, tmpH);
    conv_wmma<<<NVOX / 16, 128, 0, stream>>>(tmpH, pc2, eb2, nullptr, tmpF, nullptr, 0);
    conv_wmma<<<NVOX / 16, 128, 0, stream>>>(tmpF, pc3, eb3, tmpH, nullptr, fcv, 1);

    fused_gather_mlp<<<P_TOTAL / BM, 256, 0, stream>>>(
        fa, fcv, pix, wp, biasB, (float*)d_out);
}
